// EdgePredictionGNN_17944373362958
// MI455X (gfx1250) — compile-verified, matrix operand order checked by reading, not run on hardware
//
#include <hip/hip_runtime.h>

// ---------------------------------------------------------------------------
// EdgePredictionGNN forward for MI455X (gfx1250, wave32, WMMA).
//
// Features live in f16 (converted once per stage): a WMMA v16h operand pack
// is exactly two contiguous 8-half runs -> b128 loads, zero cvt in the hot
// loop. Candidates are A, the hoisted query tile is B, so each lane owns 8
// candidate distances in registers. Full tiles run guard-free; self-
// exclusion touches only the diagonal tile. The tile loop is ping-pong
// double-buffered with a peeled epilogue: the {sq, A} prefetch for tile t+1
// is unconditional straight-line code between wmma(t) and scan(t), so it
// fills the WMMA->VALU hazard slots and the scan needs no memory wait.
// ---------------------------------------------------------------------------

typedef __attribute__((ext_vector_type(16))) _Float16 v16h;
typedef __attribute__((ext_vector_type(8)))  _Float16 v8h;
typedef __attribute__((ext_vector_type(8)))  float    v8f;
typedef __attribute__((ext_vector_type(4)))  float    v4f;

#define LEAK 0.1f
#define BNEPS 1e-5f
#define KNN 4
#define HID 20
#define FINF 3.4e38f

// ISA 16-bit A-matrix 16x32 lane layout: lane L -> row m = L%16, hl = L/16;
// elements 0..7  hold K = kbase + hl*8 + {0..7}      (contiguous)
// elements 8..15 hold K = kbase + 16 + hl*8 + {0..7} (contiguous)

__device__ __forceinline__ v16h cat8(v8h a, v8h b) {
    return __builtin_shufflevector(a, b, 0, 1, 2, 3, 4, 5, 6, 7,
                                   8, 9, 10, 11, 12, 13, 14, 15);
}

// Unguarded pack (full tile, rows known valid).
template <int FDIM>
__device__ __forceinline__ v16h pack_tile_nc(const _Float16* __restrict__ X, int rbase,
                                             int kbase, int lane) {
    int m = lane & 15, hl = lane >> 4;
    const _Float16* p = X + (rbase + m) * FDIM;
    int k0 = kbase + hl * 8;
    int k1 = kbase + 16 + hl * 8;
    v8h a = {}, b = {};
    if (k0 + 8 <= FDIM) a = *(const v8h*)(p + k0);   // folds at compile time
    if (k1 + 8 <= FDIM) b = *(const v8h*)(p + k1);   // (FDIM=16: pad half stays 0)
    return cat8(a, b);
}

// Row-guarded pack (tail / query-tile staging).
template <int FDIM>
__device__ __forceinline__ v16h pack_tile_h(const _Float16* __restrict__ X, int rbase,
                                            int nrows, int kbase, int lane) {
    int m = lane & 15, hl = lane >> 4;
    int row = rbase + m;
    v8h a = {}, b = {};
    if (row < nrows) {
        const _Float16* p = X + row * FDIM;
        int k0 = kbase + hl * 8;
        int k1 = kbase + 16 + hl * 8;
        if (k0 + 8 <= FDIM) a = *(const v8h*)(p + k0);
        if (k1 + 8 <= FDIM) b = *(const v8h*)(p + k1);
    }
    return cat8(a, b);
}

struct Top4 { float k0, k1, k2, k3; int j0, j1, j2, j3; };

// stable compare-swap: smaller key first; strict '<' preserves insertion order
__device__ __forceinline__ void cswap(float& ka, int& ja, float& kb, int& jb) {
    if (kb < ka) { float tk = ka; ka = kb; kb = tk; int tj = ja; ja = jb; jb = tj; }
}
// lexicographic (key, idx) compare-swap for the exact cross-half merge
__device__ __forceinline__ void cswapL(float& ka, int& ja, float& kb, int& jb) {
    if (kb < ka || (kb == ka && jb < ja)) {
        float tk = ka; ka = kb; kb = tk; int tj = ja; ja = jb; jb = tj;
    }
}

__device__ __forceinline__ void top4_insert(Top4& tp, float key, int j) {
    tp.k3 = key; tp.j3 = j;
    cswap(tp.k2, tp.j2, tp.k3, tp.j3);
    cswap(tp.k1, tp.j1, tp.k2, tp.j2);
    cswap(tp.k0, tp.j0, tp.k1, tp.j1);
}

template <bool CHECK>
__device__ __forceinline__ void scan8(const v8f& acc, v4f sa, v4f sb, int jb, int qrow,
                                      Top4& tp) {
#pragma unroll
    for (int r = 0; r < 8; ++r) {
        int   j   = jb + r;
        float sqj = (r < 4) ? sa[r] : sb[r - 4];
        float key = fmaf(-2.0f, acc[r], sqj);        // ordering == full sq-distance
        bool  ok  = !CHECK || (j != qrow);
        if (ok && key < tp.k3) top4_insert(tp, key, j);   // strict '<': ties keep earlier j
    }
}

// Ping-pong double-buffered tile loop with peeled epilogue: the in-loop
// prefetch for tile t+1 is unconditional, sitting between wmma(t) and
// scan(t) where it fills the WMMA->VALU hazard slots.
template <int FDIM, bool CHECK, int CH>
__device__ __forceinline__ void knn_range(const _Float16* __restrict__ X,
                                          const float* __restrict__ sq,
                                          const v16h (&B)[CH],
                                          int t0, int t1, int lane, int qrow, Top4& tp) {
    if (t0 >= t1) return;
    int hl = lane >> 4;
    v16h A0[CH], A1[CH];
    v4f sa0, sb0, sa1, sb1;
    {   // prologue: group(t0) -> buffer 0 (sq first, then A)
        int jb = t0 * 16 + 8 * hl;
        sa0 = *(const v4f*)(sq + jb);
        sb0 = *(const v4f*)(sq + jb + 4);
#pragma unroll
        for (int c = 0; c < CH; ++c) A0[c] = pack_tile_nc<FDIM>(X, t0 * 16, 32 * c, lane);
    }
    int t = t0;
    if (t1 - t0 > 1) {
        while (true) {
            {   // consume buffer 0 (tile t); t+1 < t1 guaranteed here
                v8f acc = {};
#pragma unroll
                for (int c = 0; c < CH; ++c)
                    acc = __builtin_amdgcn_wmma_f32_16x16x32_f16(
                        false, A0[c], false, B[c], (short)0, acc, false, false);
                int jb = (t + 1) * 16 + 8 * hl;      // unconditional prefetch -> buffer 1
                sa1 = *(const v4f*)(sq + jb);
                sb1 = *(const v4f*)(sq + jb + 4);
#pragma unroll
                for (int c = 0; c < CH; ++c)
                    A1[c] = pack_tile_nc<FDIM>(X, (t + 1) * 16, 32 * c, lane);
                scan8<CHECK>(acc, sa0, sb0, t * 16 + 8 * hl, qrow, tp);
                ++t;
            }
            if (t + 1 >= t1) {                       // epilogue: last tile is in buffer 1
                v8f acc = {};
#pragma unroll
                for (int c = 0; c < CH; ++c)
                    acc = __builtin_amdgcn_wmma_f32_16x16x32_f16(
                        false, A1[c], false, B[c], (short)0, acc, false, false);
                scan8<CHECK>(acc, sa1, sb1, t * 16 + 8 * hl, qrow, tp);
                return;
            }
            {   // consume buffer 1 (tile t); t+1 < t1 guaranteed here
                v8f acc = {};
#pragma unroll
                for (int c = 0; c < CH; ++c)
                    acc = __builtin_amdgcn_wmma_f32_16x16x32_f16(
                        false, A1[c], false, B[c], (short)0, acc, false, false);
                int jb = (t + 1) * 16 + 8 * hl;      // unconditional prefetch -> buffer 0
                sa0 = *(const v4f*)(sq + jb);
                sb0 = *(const v4f*)(sq + jb + 4);
#pragma unroll
                for (int c = 0; c < CH; ++c)
                    A0[c] = pack_tile_nc<FDIM>(X, (t + 1) * 16, 32 * c, lane);
                scan8<CHECK>(acc, sa1, sb1, t * 16 + 8 * hl, qrow, tp);
                ++t;
            }
            if (t + 1 >= t1) {                       // epilogue: last tile is in buffer 0
                v8f acc = {};
#pragma unroll
                for (int c = 0; c < CH; ++c)
                    acc = __builtin_amdgcn_wmma_f32_16x16x32_f16(
                        false, A0[c], false, B[c], (short)0, acc, false, false);
                scan8<CHECK>(acc, sa0, sb0, t * 16 + 8 * hl, qrow, tp);
                return;
            }
        }
    } else {    // single tile: no prefetch
        v8f acc = {};
#pragma unroll
        for (int c = 0; c < CH; ++c)
            acc = __builtin_amdgcn_wmma_f32_16x16x32_f16(
                false, A0[c], false, B[c], (short)0, acc, false, false);
        scan8<CHECK>(acc, sa0, sb0, t * 16 + 8 * hl, qrow, tp);
    }
}

// ---------------------------------------------------------------------------
// f32 -> f16 feature conversion + squared norms (of the f16-rounded values,
// so kNN keys are self-consistent with the WMMA dot products).
// ---------------------------------------------------------------------------
__global__ void cvt_sq_kernel(const float* __restrict__ X, _Float16* __restrict__ Xh,
                              float* __restrict__ sq, int n, int f) {
    int i = blockIdx.x * blockDim.x + threadIdx.x;
    if (i >= n) return;
    float s = 0.0f;
    for (int c = 0; c < f; c += 8) {
        v4f a = *(const v4f*)(X + i * f + c);
        v4f b = *(const v4f*)(X + i * f + c + 4);
        v8h h;
#pragma unroll
        for (int t = 0; t < 4; ++t) { h[t] = (_Float16)a[t]; h[4 + t] = (_Float16)b[t]; }
#pragma unroll
        for (int t = 0; t < 8; ++t) { float v = (float)h[t]; s += v * v; }
        *(v8h*)(Xh + i * f + c) = h;
    }
    sq[i] = s;
}

__global__ void sqnorm_h_kernel(const _Float16* __restrict__ Xh, float* __restrict__ sq,
                                int n, int f) {
    int i = blockIdx.x * blockDim.x + threadIdx.x;
    if (i >= n) return;
    float s = 0.0f;
    for (int c = 0; c < f; c += 8) {
        v8h h = *(const v8h*)(Xh + i * f + c);
#pragma unroll
        for (int t = 0; t < 8; ++t) { float v = (float)h[t]; s += v * v; }
    }
    sq[i] = s;
}

__global__ void zero_stats_kernel(float* __restrict__ stats) {
    if (threadIdx.x < 2 * HID) stats[threadIdx.x] = 0.0f;
}

// ---------------------------------------------------------------------------
// Fused kNN. One wave per 16 queries; B (queries) hoisted. Self-exclusion
// touches only the diagonal tile; full tiles run guard-free.
// ---------------------------------------------------------------------------
template <int FDIM, bool INCLUDE_SELF>
__global__ void knn_kernel(const _Float16* __restrict__ X, const float* __restrict__ sq,
                           int* __restrict__ idx, int n) {
    constexpr int CH = (FDIM + 31) / 32;
    __shared__ float mk[32 * KNN];
    __shared__ int   mi[32 * KNN];

    int lane  = threadIdx.x & 31;
    int qbase = blockIdx.x * 16;
    int nq    = lane & 15;              // query column owned by this lane
    int hl    = lane >> 4;
    int qrow  = qbase + nq;

    v16h B[CH];
#pragma unroll
    for (int c = 0; c < CH; ++c) B[c] = pack_tile_h<FDIM>(X, qbase, n, 32 * c, lane);

    Top4 tp = {FINF, FINF, FINF, FINF, 0, 0, 0, 0};

    int nfull = n >> 4;                 // full 16-candidate tiles
    int diag  = blockIdx.x;             // tile that can contain j == qrow
    if (INCLUDE_SELF || diag >= nfull) {
        knn_range<FDIM, false, CH>(X, sq, B, 0, nfull, lane, qrow, tp);
    } else {
        knn_range<FDIM, false, CH>(X, sq, B, 0, diag, lane, qrow, tp);
        knn_range<FDIM, true,  CH>(X, sq, B, diag, diag + 1, lane, qrow, tp);
        knn_range<FDIM, false, CH>(X, sq, B, diag + 1, nfull, lane, qrow, tp);
    }
    if (n & 15) {                       // guarded tail tile (dead for N=12000)
        int cbase = nfull * 16;
        v8f acc = {};
#pragma unroll
        for (int c = 0; c < CH; ++c) {
            v16h A = pack_tile_h<FDIM>(X, cbase, n, 32 * c, lane);
            acc = __builtin_amdgcn_wmma_f32_16x16x32_f16(
                false, A, false, B[c], (short)0, acc, false, false);
        }
        int jb = cbase + 8 * hl;
#pragma unroll
        for (int r = 0; r < 8; ++r) {
            int j = jb + r;
            if (j < n && (INCLUDE_SELF || j != qrow)) {
                float key = fmaf(-2.0f, acc[r], sq[j]);
                if (key < tp.k3) top4_insert(tp, key, j);
            }
        }
    }

    // merge the two half-lists per query (bitonic lowest-4-of-8, exact lex order)
    mk[lane * 4 + 0] = tp.k0; mk[lane * 4 + 1] = tp.k1;
    mk[lane * 4 + 2] = tp.k2; mk[lane * 4 + 3] = tp.k3;
    mi[lane * 4 + 0] = tp.j0; mi[lane * 4 + 1] = tp.j1;
    mi[lane * 4 + 2] = tp.j2; mi[lane * 4 + 3] = tp.j3;
    __syncthreads();
    if (lane < 16 && qrow < n) {
        int p = (lane + 16) * 4;
        float b0 = mk[p + 0], b1 = mk[p + 1], b2 = mk[p + 2], b3 = mk[p + 3];
        int   c0 = mi[p + 0], c1 = mi[p + 1], c2 = mi[p + 2], c3 = mi[p + 3];
        // bitonic split: a_i vs b_{3-i} -> a holds the 4 smallest
        cswapL(tp.k0, tp.j0, b3, c3); cswapL(tp.k1, tp.j1, b2, c2);
        cswapL(tp.k2, tp.j2, b1, c1); cswapL(tp.k3, tp.j3, b0, c0);
        // sort the (bitonic) low half
        cswapL(tp.k0, tp.j0, tp.k2, tp.j2); cswapL(tp.k1, tp.j1, tp.k3, tp.j3);
        cswapL(tp.k0, tp.j0, tp.k1, tp.j1); cswapL(tp.k2, tp.j2, tp.k3, tp.j3);
        idx[qrow * KNN + 0] = tp.j0; idx[qrow * KNN + 1] = tp.j1;
        idx[qrow * KNN + 2] = tp.j2; idx[qrow * KNN + 3] = tp.j3;
    }
}

// ---------------------------------------------------------------------------
// Edge MLP first layer via WMMA. MODE 0: e=[x_i, x_j-x_i]; MODE 1: e=[h_src,h_dst].
// Each 8-half pack segment lies entirely inside one half of the concat, so
// segments are single b128 loads (packed f16 subtract for the x_j-x_i half).
// ---------------------------------------------------------------------------
template <int F, int MODE>
__device__ __forceinline__ v8h seg_edge_h(const _Float16* __restrict__ Xi,
                                          const _Float16* __restrict__ Xj, int ks) {
    if (ks + 8 <= F) {
        return (MODE == 0) ? *(const v8h*)(Xi + ks)
                           : *(const v8h*)(Xj + ks);     // h_src first for link head
    } else {
        int kk = ks - F;
        if (MODE == 0) return *(const v8h*)(Xj + kk) - *(const v8h*)(Xi + kk);
        else           return *(const v8h*)(Xi + kk);
    }
}

template <int F, int MODE>
__device__ __forceinline__ v16h packA_edge(const _Float16* __restrict__ X,
                                           const int* __restrict__ idxbuf,
                                           int ebase, int nedge, int kbase, int lane) {
    int m = lane & 15, hl = lane >> 4;
    int eidx = ebase + m;
    if (eidx >= nedge) eidx = nedge - 1;
    int i = eidx >> 2;                   // center node (dst)
    int j = idxbuf[eidx];                // neighbor (src)
    const _Float16* Xi = X + i * F;
    const _Float16* Xj = X + j * F;
    v8h a = seg_edge_h<F, MODE>(Xi, Xj, kbase + hl * 8);
    v8h b = seg_edge_h<F, MODE>(Xi, Xj, kbase + 16 + hl * 8);
    return cat8(a, b);
}

__device__ __forceinline__ v16h packB_W1(const float* __restrict__ W1, int nin,
                                         int ntile, int kbase, int lane) {
    int u = ntile * 16 + (lane & 15), hl = lane >> 4;
    v16h v = {};
    if (u < HID) {
        const float* p = W1 + u * nin;
        int k0 = kbase + hl * 8, k1 = kbase + 16 + hl * 8;
        v4f a0 = *(const v4f*)(p + k0), a1 = *(const v4f*)(p + k0 + 4);
        v4f b0 = *(const v4f*)(p + k1), b1 = *(const v4f*)(p + k1 + 4);
#pragma unroll
        for (int t = 0; t < 4; ++t) {
            v[t]      = (_Float16)a0[t]; v[4 + t]  = (_Float16)a1[t];
            v[8 + t]  = (_Float16)b0[t]; v[12 + t] = (_Float16)b1[t];
        }
    }
    return v;
}

template <int F, int MODE>
__global__ void edge_hidden_kernel(const _Float16* __restrict__ X, const int* __restrict__ idxbuf,
                                   const float* __restrict__ W1, const float* __restrict__ b1,
                                   float* __restrict__ hid, float* __restrict__ stats,
                                   int nedge) {
    constexpr int NIN = 2 * F;
    constexpr int CH  = NIN / 32;
    __shared__ float ssum[HID], ssq[HID];

    int tid = threadIdx.x;
    if (tid < HID) { ssum[tid] = 0.0f; ssq[tid] = 0.0f; }
    __syncthreads();

    int wave = tid >> 5, lane = tid & 31;
    int ebase = (blockIdx.x * 8 + wave) * 16;

    v8f acc0 = {}, acc1 = {};
#pragma unroll
    for (int c = 0; c < CH; ++c) {
        v16h A  = packA_edge<F, MODE>(X, idxbuf, ebase, nedge, 32 * c, lane);
        v16h B0 = packB_W1(W1, NIN, 0, 32 * c, lane);
        v16h B1 = packB_W1(W1, NIN, 1, 32 * c, lane);
        acc0 = __builtin_amdgcn_wmma_f32_16x16x32_f16(false, A, false, B0, (short)0, acc0, false, false);
        acc1 = __builtin_amdgcn_wmma_f32_16x16x32_f16(false, A, false, B1, (short)0, acc1, false, false);
    }

    int nn = lane & 15, hl = lane >> 4;
    float ps = 0.0f, ps2 = 0.0f, qs = 0.0f, qs2 = 0.0f;
#pragma unroll
    for (int r = 0; r < 8; ++r) {
        int eidx = ebase + r + 8 * hl;
        if (eidx >= nedge) continue;
        float v0 = acc0[r] + b1[nn];                 // units 0..15
        hid[eidx * HID + nn] = v0;
        ps += v0; ps2 += v0 * v0;
        if (nn < HID - 16) {                         // units 16..19
            float v1 = acc1[r] + b1[16 + nn];
            hid[eidx * HID + 16 + nn] = v1;
            qs += v1; qs2 += v1 * v1;
        }
    }
    atomicAdd(&ssum[nn], ps);  atomicAdd(&ssq[nn], ps2);
    if (nn < HID - 16) { atomicAdd(&ssum[16 + nn], qs); atomicAdd(&ssq[16 + nn], qs2); }
    __syncthreads();
    if (tid < HID) { atomicAdd(&stats[tid], ssum[tid]); atomicAdd(&stats[HID + tid], ssq[tid]); }
}

// ---------------------------------------------------------------------------
// BN(finalize) + LeakyReLU + Linear(20->64) + max over K. Thread = (node, o).
// Output written directly as f16 (next stage consumes f16 anyway).
// ---------------------------------------------------------------------------
__global__ void conv_out_kernel(const float* __restrict__ hid, const float* __restrict__ stats,
                                const float* __restrict__ g, const float* __restrict__ be,
                                const float* __restrict__ W2, const float* __restrict__ b2,
                                _Float16* __restrict__ out, int nnode, float invM) {
    __shared__ float sW2[64 * HID];
    __shared__ float sc[HID], sh[HID];
    int tid = threadIdx.x;
    for (int i = tid; i < 64 * HID; i += blockDim.x) sW2[i] = W2[i];
    if (tid < HID) {
        float mu  = stats[tid] * invM;
        float var = stats[HID + tid] * invM - mu * mu;     // biased var
        float s   = g[tid] * rsqrtf(var + BNEPS);
        sc[tid] = s; sh[tid] = be[tid] - s * mu;
    }
    __syncthreads();
    int gidx = blockIdx.x * blockDim.x + tid;
    if (gidx >= nnode * 64) return;
    int node = gidx >> 6, o = gidx & 63;
    float best = -FINF;
#pragma unroll
    for (int k = 0; k < KNN; ++k) {
        int eidx = node * KNN + k;
        float acc = b2[o];
#pragma unroll
        for (int q = 0; q < HID / 4; ++q) {
            v4f hv = *(const v4f*)(hid + eidx * HID + 4 * q);
#pragma unroll
            for (int t = 0; t < 4; ++t) {
                int u = 4 * q + t;
                float y = sc[u] * hv[t] + sh[u];
                y = (y >= 0.0f) ? y : LEAK * y;
                acc += sW2[o * HID + u] * y;
            }
        }
        best = fmaxf(best, acc);
    }
    out[node * 64 + o] = (_Float16)best;
}

// ---------------------------------------------------------------------------
// Link head: BN + LeakyReLU + Linear(20->1) + sigmoid, plus edge_index.
// d_out layout: probs[nedge] (f32) then edge_index[2*nedge] (i32 bits).
// ---------------------------------------------------------------------------
__global__ void lp_out_kernel(const float* __restrict__ hid, const float* __restrict__ stats,
                              const float* __restrict__ g, const float* __restrict__ be,
                              const float* __restrict__ W2, const float* __restrict__ b2,
                              const int* __restrict__ idxbuf,
                              float* __restrict__ probs, int* __restrict__ ei,
                              int nedge, float invM) {
    __shared__ float sc[HID], sh[HID], sW[HID];
    int tid = threadIdx.x;
    if (tid < HID) {
        float mu  = stats[tid] * invM;
        float var = stats[HID + tid] * invM - mu * mu;
        float s   = g[tid] * rsqrtf(var + BNEPS);
        sc[tid] = s; sh[tid] = be[tid] - s * mu;
        sW[tid] = W2[tid];
    }
    __syncthreads();
    int e = blockIdx.x * blockDim.x + tid;
    if (e >= nedge) return;
    float acc = b2[0];
#pragma unroll
    for (int q = 0; q < HID / 4; ++q) {
        v4f hv = *(const v4f*)(hid + e * HID + 4 * q);
#pragma unroll
        for (int t = 0; t < 4; ++t) {
            int u = 4 * q + t;
            float y = sc[u] * hv[t] + sh[u];
            y = (y >= 0.0f) ? y : LEAK * y;
            acc += sW[u] * y;
        }
    }
    probs[e] = 1.0f / (1.0f + __expf(-acc));
    ei[e]         = idxbuf[e];   // row 0: src (neighbor j)
    ei[nedge + e] = e >> 2;      // row 1: dst (center i)
}

// ---------------------------------------------------------------------------
// Host-side launch
// ---------------------------------------------------------------------------
extern "C" void kernel_launch(void* const* d_in, const int* in_sizes, int n_in,
                              void* d_out, int out_size, void* d_ws, size_t ws_size,
                              hipStream_t stream) {
    (void)n_in; (void)out_size; (void)ws_size;
    const float* x     = (const float*)d_in[0];
    const float* c1_W1 = (const float*)d_in[1];
    const float* c1_b1 = (const float*)d_in[2];
    const float* c1_g  = (const float*)d_in[3];
    const float* c1_be = (const float*)d_in[4];
    const float* c1_W2 = (const float*)d_in[5];
    const float* c1_b2 = (const float*)d_in[6];
    const float* c2_W1 = (const float*)d_in[7];
    const float* c2_b1 = (const float*)d_in[8];
    const float* c2_g  = (const float*)d_in[9];
    const float* c2_be = (const float*)d_in[10];
    const float* c2_W2 = (const float*)d_in[11];
    const float* c2_b2 = (const float*)d_in[12];
    const float* lp_W1 = (const float*)d_in[13];
    const float* lp_b1 = (const float*)d_in[14];
    const float* lp_g  = (const float*)d_in[15];
    const float* lp_be = (const float*)d_in[16];
    const float* lp_W2 = (const float*)d_in[17];
    const float* lp_b2 = (const float*)d_in[18];

    const int n     = in_sizes[0] / 16;   // 12000
    const int nedge = n * KNN;            // 48000
    const float invM = 1.0f / (float)nedge;

    // workspace layout (256B aligned slabs)
    char* ws = (char*)d_ws;
    size_t off = 0;
    auto alloc = [&](size_t bytes) { void* p = ws + off; off = (off + bytes + 255) & ~(size_t)255; return p; };
    _Float16* xh    = (_Float16*)alloc((size_t)n * 16 * sizeof(_Float16));
    _Float16* h1h   = (_Float16*)alloc((size_t)n * 64 * sizeof(_Float16));
    _Float16* h2h   = (_Float16*)alloc((size_t)n * 64 * sizeof(_Float16));
    float*    sq    = (float*)alloc((size_t)n * sizeof(float));
    int*      idxb  = (int*)  alloc((size_t)n * KNN * sizeof(int));
    float*    hid   = (float*)alloc((size_t)nedge * HID * sizeof(float));
    float*    stats = (float*)alloc(2 * HID * sizeof(float));

    float* probs = (float*)d_out;
    int*   ei    = (int*)((float*)d_out + nedge);

    const int qtiles  = (n + 15) / 16;
    const int eblocks = (nedge + 127) / 128;
    const int nblk    = (n + 255) / 256;

    // ---- EdgeConv 1: kNN(x, 16-dim, self) -> MLP -> max-agg -> h1 (f16) ----
    cvt_sq_kernel<<<nblk, 256, 0, stream>>>(x, xh, sq, n, 16);
    knn_kernel<16, true><<<qtiles, 32, 0, stream>>>(xh, sq, idxb, n);
    zero_stats_kernel<<<1, 64, 0, stream>>>(stats);
    edge_hidden_kernel<16, 0><<<eblocks, 256, 0, stream>>>(xh, idxb, c1_W1, c1_b1, hid, stats, nedge);
    conv_out_kernel<<<(n * 64 + 255) / 256, 256, 0, stream>>>(hid, stats, c1_g, c1_be, c1_W2, c1_b2, h1h, n, invM);

    // ---- EdgeConv 2: kNN(h1, 64-dim, self) -> MLP -> max-agg -> h2 (f16) ----
    sqnorm_h_kernel<<<nblk, 256, 0, stream>>>(h1h, sq, n, 64);
    knn_kernel<64, true><<<qtiles, 32, 0, stream>>>(h1h, sq, idxb, n);
    zero_stats_kernel<<<1, 64, 0, stream>>>(stats);
    edge_hidden_kernel<64, 0><<<eblocks, 256, 0, stream>>>(h1h, idxb, c2_W1, c2_b1, hid, stats, nedge);
    conv_out_kernel<<<(n * 64 + 255) / 256, 256, 0, stream>>>(hid, stats, c2_g, c2_be, c2_W2, c2_b2, h2h, n, invM);

    // ---- Link prediction: kNN(h2, no self) -> MLP([h_src,h_dst]) -> sigmoid ----
    sqnorm_h_kernel<<<nblk, 256, 0, stream>>>(h2h, sq, n, 64);
    knn_kernel<64, false><<<qtiles, 32, 0, stream>>>(h2h, sq, idxb, n);
    zero_stats_kernel<<<1, 64, 0, stream>>>(stats);
    edge_hidden_kernel<64, 1><<<eblocks, 256, 0, stream>>>(h2h, idxb, lp_W1, lp_b1, hid, stats, nedge);
    lp_out_kernel<<<(nedge + 255) / 256, 256, 0, stream>>>(hid, stats, lp_g, lp_be, lp_W2, lp_b2,
                                                           idxb, probs, ei, nedge, invM);
}
